// CPABTransform2D_23038204576189
// MI455X (gfx1250) — matrix operand order repaired and checked by reading, not compiled, not run on hardware
//
#include <hip/hip_runtime.h>
#include <hip/hip_bf16.h>

// ---------------- problem constants ----------------
#define NCELLS 32
#define DTHETA 31
#define HID    256
#define NPTS   524288
#define EPSF   1e-7f
#define TINYF  1e-10f

typedef _Float16 v16h __attribute__((ext_vector_type(16)));
typedef _Float16 v8h  __attribute__((ext_vector_type(8)));
typedef float    v8f  __attribute__((ext_vector_type(8)));

union V16 { v16h v; v8h h[2]; };

// ---------------- workspace layout (bytes) ----------------
#define WS_W2T 0                        // f16 [256 n][256 k]  (transposed w2)
#define WS_AWT (WS_W2T + 256*256*2)     // f16 [64 m][256 k]   (w3 @ B^T, transposed)
#define WS_U   (WS_AWT + 64*256*2)      // f32 [256]
#define WS_V   (WS_U + 256*4)           // f32 [256]
#define WS_AB  (WS_V + 256*4)           // f32 [64]

// =====================================================================
// prep 1: fuse layers 0+1 into affine-in-x1:  u = w0@w1 , v = b0@w1 + b1
// =====================================================================
__global__ __launch_bounds__(256) void prep_fuse(
    const float* __restrict__ w0, const float* __restrict__ b0,
    const float* __restrict__ w1, const float* __restrict__ b1,
    float* __restrict__ U, float* __restrict__ V)
{
    const int j = threadIdx.x;
    float uj = 0.f, vj = 0.f;
    for (int i = 0; i < HID; ++i) {
        const float w = w1[i * HID + j];
        uj = fmaf(w0[i], w, uj);
        vj = fmaf(b0[i], w, vj);
    }
    U[j] = uj;
    V[j] = vj + b1[j];
}

// =====================================================================
// prep 2: w2 -> f16, transposed to [n][k] so WMMA B-frags are contiguous
// =====================================================================
__global__ __launch_bounds__(256) void prep_w2(
    const float* __restrict__ w2, _Float16* __restrict__ W2t)
{
    const int n = blockIdx.x;        // output column of GEMM1
    const int k = threadIdx.x;       // reduction index
    W2t[n * HID + k] = (_Float16)w2[k * HID + n];
}

// =====================================================================
// prep 3: fold w3 @ Bhat^T (hat-function null-space basis of the CPA
// constraints: knot d+1 -> cell d: a=+nc,b=-d ; cell d+1: a=-nc,b=d+2)
// stored f16 transposed [m][k];  Ab = b3 @ Bhat^T  (f32[64])
// =====================================================================
__global__ __launch_bounds__(256) void prep_w3(
    const float* __restrict__ w3, const float* __restrict__ b3,
    _Float16* __restrict__ AWt, float* __restrict__ AB)
{
    const int m = blockIdx.x;        // 0..63 : (cell c, comp) flattened
    const int h = threadIdx.x;       // 0..255
    const int c    = m >> 1;
    const int comp = m & 1;          // 0 = slope a, 1 = offset b
    float val = 0.f;
    if (c <= DTHETA - 1) val += w3[h * DTHETA + c]     * (comp ? -(float)c       : 32.0f);
    if (c >= 1)          val += w3[h * DTHETA + (c-1)] * (comp ?  (float)(c + 1) : -32.0f);
    AWt[m * HID + h] = (_Float16)val;
    if (h == 0) {
        float ab = 0.f;
        if (c <= DTHETA - 1) ab += b3[c]     * (comp ? -(float)c       : 32.0f);
        if (c >= 1)          ab += b3[c - 1] * (comp ?  (float)(c + 1) : -32.0f);
        AB[m] = ab;
    }
}

// =====================================================================
// main fused kernel: analytic h1 -> WMMA GEMM1 (256x256) -> relu ->
// WMMA GEMM2 (256x64) -> per-point closed-form CPAB ODE integration
// 256 threads = 8 wave32, 128 rows/block, 4096 blocks
// Async global->LDS weight staging; 4-way interleaved WMMA chains for
// intra-wave ILP (occupancy is LDS-limited to 2 waves/SIMD).
// =====================================================================
__global__ __launch_bounds__(256) void cpab_main(
    const float* __restrict__ x,  const float* __restrict__ b2,
    const _Float16* __restrict__ w2t, const _Float16* __restrict__ awt,
    const float* __restrict__ u,  const float* __restrict__ v,
    const float* __restrict__ ab, float* __restrict__ out)
{
    __shared__ struct {
        alignas(16) _Float16 w2t[256 * 256];                          // 128 KB
        alignas(16) _Float16 awt[64 * 256];                           //  32 KB
        union alignas(16) PW { _Float16 h2[16 * 256]; float ac[16 * 64]; } pw[8]; // 64 KB
        float u[256], vv[256], b2[256], ab[64];                       // ~3.3 KB
    } sm;

    const int tid = threadIdx.x;

    // ---- async staging of f16 weights: global -> LDS, no VGPR round-trip ----
    {
        const unsigned ldsW2 = (unsigned)(uintptr_t)(&sm.w2t[0]);
        for (int i = tid; i < (256 * 256 * 2) / 16; i += 256) {
            const unsigned lo = ldsW2 + (unsigned)i * 16u;
            const unsigned go = (unsigned)i * 16u;
            asm volatile("global_load_async_to_lds_b128 %0, %1, %2"
                         :: "v"(lo), "v"(go), "s"(w2t) : "memory");
        }
        const unsigned ldsAW = (unsigned)(uintptr_t)(&sm.awt[0]);
        for (int i = tid; i < (64 * 256 * 2) / 16; i += 256) {
            const unsigned lo = ldsAW + (unsigned)i * 16u;
            const unsigned go = (unsigned)i * 16u;
            asm volatile("global_load_async_to_lds_b128 %0, %1, %2"
                         :: "v"(lo), "v"(go), "s"(awt) : "memory");
        }
        sm.u[tid]  = u[tid];
        sm.vv[tid] = v[tid];
        sm.b2[tid] = b2[tid];
        if (tid < 64) sm.ab[tid] = ab[tid];
        asm volatile("s_wait_asynccnt 0" ::: "memory");
    }
    __syncthreads();

    const int wave = tid >> 5;
    const int lane = tid & 31;
    const int lr   = lane & 15;     // row within 16-row tile / N within col tile
    const int hi   = lane >> 4;     // half-wave select
    const int row0 = blockIdx.x * 128 + wave * 16;

    // ---- analytic layer-1: h1 = relu(x1*u + v), built as 8 A-fragments ----
    const float x1c = fminf(fmaxf(x[2 * (row0 + lr) + 1], EPSF), 1.f - EPSF);
    V16 aF[8];
#pragma unroll
    for (int kc = 0; kc < 8; ++kc) {
        const int kb1 = kc * 32 + hi * 8;       // ISA 16-bit A layout
        const int kb2 = kb1 + 16;
#pragma unroll
        for (int j = 0; j < 8; ++j) {
            aF[kc].v[j]     = (_Float16)fmaxf(fmaf(x1c, sm.u[kb1 + j], sm.vv[kb1 + j]), 0.f);
            aF[kc].v[8 + j] = (_Float16)fmaxf(fmaf(x1c, sm.u[kb2 + j], sm.vv[kb2 + j]), 0.f);
        }
    }

    _Float16* h2b = sm.pw[wave].h2;

    // ---- GEMM1: h2 = relu(h1 @ w2 + b2) ----
    // 4 independent accumulation chains per pass to hide the WMMA D->C
    // RAW latency (only 2 waves/SIMD at this LDS footprint).
    for (int nt = 0; nt < 16; nt += 4) {
        v8f acc[4] = {};
#pragma unroll
        for (int kc = 0; kc < 8; ++kc) {
            const int kb = kc * 32 + hi * 8;    // B: ISA 16-bit layout (matches A)
#pragma unroll
            for (int t = 0; t < 4; ++t) {
                const int ncol = (nt + t) * 16 + lr;
                V16 bF;
                bF.h[0] = *(const v8h*)&sm.w2t[ncol * 256 + kb];
                bF.h[1] = *(const v8h*)&sm.w2t[ncol * 256 + kb + 16];
                acc[t] = __builtin_amdgcn_wmma_f32_16x16x32_f16(
                    false, aF[kc].v, false, bF.v, (short)0, acc[t], false, false);
            }
        }
#pragma unroll
        for (int t = 0; t < 4; ++t) {
            const int ncol = (nt + t) * 16 + lr;
            const float bias = sm.b2[ncol];
#pragma unroll
            for (int i = 0; i < 8; ++i) {
                const int r = i + hi * 8;       // C-layout row
                h2b[r * 256 + ncol] = (_Float16)fmaxf(acc[t][i] + bias, 0.f);
            }
        }
    }
    asm volatile("s_wait_dscnt 0" ::: "memory");  // wave's h2 stores visible to its own loads

    // ---- GEMM2: Acoef = h2 @ (w3@B^T) + b3@B^T, all 4 N-tiles interleaved ----
    v8f acc2[4] = {};
#pragma unroll
    for (int kc = 0; kc < 8; ++kc) {
        V16 aG;
        const int kb1 = kc * 32 + hi * 8;
        aG.h[0] = *(const v8h*)&h2b[lr * 256 + kb1];
        aG.h[1] = *(const v8h*)&h2b[lr * 256 + kb1 + 16];
        const int kb = kc * 32 + hi * 8;
#pragma unroll
        for (int t = 0; t < 4; ++t) {
            const int m = t * 16 + lr;
            V16 bF;
            bF.h[0] = *(const v8h*)&sm.awt[m * 256 + kb];
            bF.h[1] = *(const v8h*)&sm.awt[m * 256 + kb + 16];
            acc2[t] = __builtin_amdgcn_wmma_f32_16x16x32_f16(
                false, aG.v, false, bF.v, (short)0, acc2[t], false, false);
        }
    }
    // spill per-row A coefficients (reuses h2 LDS region via union)
    float* acb = sm.pw[wave].ac;
#pragma unroll
    for (int t = 0; t < 4; ++t) {
        const int m    = t * 16 + lr;
        const float a0 = sm.ab[m];
#pragma unroll
        for (int i = 0; i < 8; ++i) {
            const int r = i + hi * 8;
            acb[r * 64 + m] = acc2[t][i] + a0;
        }
    }
    __syncthreads();

    // ---- closed-form CPAB integration, one thread per row ----
    if (tid < 128) {
        const int row = blockIdx.x * 128 + tid;
        const float* Ac = sm.pw[tid >> 4].ac + (tid & 15) * 64;
        float xc   = fminf(fmaxf(x[2 * row + 0], EPSF), 1.f - EPSF);
        float x1o  = fminf(fmaxf(x[2 * row + 1], EPSF), 1.f - EPSF);
        float trem = 1.f, logg = 0.f;
        for (int it = 0; it < NCELLS + 2; ++it) {
            int c = (int)floorf(xc * 32.f);
            c = c < 0 ? 0 : (c > 31 ? 31 : c);
            const float a = Ac[2 * c];
            const float b = Ac[2 * c + 1];
            const float vel  = fmaf(a, xc, b);
            const float xbnd = (vel >= 0.f ? (float)(c + 1) : (float)c) * (1.f / 32.f);
            const bool  small_a = fabsf(a) < TINYF;
            const float safe_a  = small_a ? 1.f : a;
            const float e     = b / safe_a;
            const float denom = xc + e;
            const float sden  = fabsf(denom) < TINYF ? (denom >= 0.f ? TINYF : -TINYF) : denom;
            const float ratio = (xbnd + e) / sden;
            const float th_e  = __logf(fmaxf(ratio, TINYF)) / safe_a;
            const float sb    = fabsf(b) < TINYF ? TINYF : b;
            const float th_l  = (xbnd - xc) / sb;
            const float inf   = __builtin_inff();
            float thit = small_a ? th_l : (ratio > TINYF ? th_e : inf);
            thit = thit <= 0.f ? inf : thit;
            const float dt    = fminf(thit, trem);
            const bool  cross = thit <= trem;
            const float adt   = a * dt;
            const float x_e   = fmaf(xc + e, __expf(adt), -e);
            const float x_l   = fmaf(b, dt, xc);
            float xn = small_a ? x_l : x_e;
            xn = cross ? xbnd + (vel >= 0.f ? 1e-7f : -1e-7f) : xn;
            xn = fminf(fmaxf(xn, EPSF), 1.f - EPSF);
            logg += adt;               // log(prod exp(a*dt)) computed exactly
            trem -= dt;
            xc = xn;
        }
        out[2 * row + 0]            = xc;     // z2
        out[2 * row + 1]            = x1o;    // z second component = x1
        out[2 * NPTS + 2 * row + 0] = logg;   // log dphi/dx
        out[2 * NPTS + 2 * row + 1] = 0.f;
    }
}

// =====================================================================
extern "C" void kernel_launch(void* const* d_in, const int* in_sizes, int n_in,
                              void* d_out, int out_size, void* d_ws, size_t ws_size,
                              hipStream_t stream)
{
    (void)in_sizes; (void)n_in; (void)out_size; (void)ws_size;
    const float* x  = (const float*)d_in[0];
    const float* w0 = (const float*)d_in[1];
    const float* b0 = (const float*)d_in[2];
    const float* w1 = (const float*)d_in[3];
    const float* b1 = (const float*)d_in[4];
    const float* w2 = (const float*)d_in[5];
    const float* b2 = (const float*)d_in[6];
    const float* w3 = (const float*)d_in[7];
    const float* b3 = (const float*)d_in[8];

    char* ws = (char*)d_ws;
    _Float16* W2t = (_Float16*)(ws + WS_W2T);
    _Float16* AWt = (_Float16*)(ws + WS_AWT);
    float*    U   = (float*)(ws + WS_U);
    float*    V   = (float*)(ws + WS_V);
    float*    AB  = (float*)(ws + WS_AB);

    prep_fuse<<<1,   256, 0, stream>>>(w0, b0, w1, b1, U, V);
    prep_w2  <<<256, 256, 0, stream>>>(w2, W2t);
    prep_w3  <<<64,  256, 0, stream>>>(w3, b3, AWt, AB);
    cpab_main<<<NPTS / 128, 256, 0, stream>>>(x, b2, W2t, AWt, U, V, AB, (float*)d_out);
}
